// AlexNet_flags_43001212567614
// MI455X (gfx1250) — matrix operand order repaired and checked by reading, not compiled, not run on hardware
//
#include <hip/hip_runtime.h>
#include <hip/hip_bf16.h>

typedef __attribute__((ext_vector_type(16))) _Float16 v16h;
typedef __attribute__((ext_vector_type(8)))  _Float16 v8h;
typedef __attribute__((ext_vector_type(8)))  float    v8f;
typedef __attribute__((ext_vector_type(4)))  int      v4i;

union HFrag { v16h v; v8h h[2]; _Float16 e[16]; };
static_assert(sizeof(HFrag) == 32, "frag size");

// CDNA5 async global->LDS copy (ASYNCcnt-tracked), feature-gated so the file
// compiles on any toolchain; falls back to plain vector LDS stores.
#if defined(__HIP_DEVICE_COMPILE__) &&                                        \
    __has_builtin(__builtin_amdgcn_global_load_async_to_lds_b128) &&          \
    __has_builtin(__builtin_amdgcn_s_wait_asynccnt)
#define USE_ASYNC_COPY 1
#else
#define USE_ASYNC_COPY 0
#endif

__device__ __forceinline__ void copy16B_g2l(const _Float16* src,
                                            _Float16* dst) {
#if USE_ASYNC_COPY
  __builtin_amdgcn_global_load_async_to_lds_b128(
      (__attribute__((address_space(1))) v4i*)(uintptr_t)src,
      (__attribute__((address_space(3))) v4i*)(uint32_t)(uintptr_t)dst, 0, 0);
#else
  *(v8h*)dst = *(const v8h*)src;
#endif
}

__device__ __forceinline__ void wait_async_copies() {
#if USE_ASYNC_COPY
  __builtin_amdgcn_s_wait_asynccnt(0);
#endif
}

// ---------------------------------------------------------------------------
// f32 -> f16 elementwise convert (weights)
// ---------------------------------------------------------------------------
__global__ void cvt_f32_f16(const float* __restrict__ in,
                            _Float16* __restrict__ out, int n) {
  int i = blockIdx.x * blockDim.x + threadIdx.x;
  if (i < n) out[i] = (_Float16)in[i];
}

// ---------------------------------------------------------------------------
// conv1: 3->64, 3x3, pad 1, 32x32, fused bias+ReLU, f32 in -> f16 out.
// K=27 is too ragged for WMMA and is <1% of total FLOPs -> VALU.
// ---------------------------------------------------------------------------
__global__ __launch_bounds__(256) void conv1_relu_f16(
    const float* __restrict__ x, const float* __restrict__ w,
    const float* __restrict__ b, _Float16* __restrict__ out) {
  int idx = blockIdx.x * 256 + threadIdx.x;      // [256][64][32][32]
  int ox = idx & 31;
  int oy = (idx >> 5) & 31;
  int co = (idx >> 10) & 63;
  int bi = idx >> 16;
  float acc = b[co];
#pragma unroll
  for (int ci = 0; ci < 3; ++ci) {
#pragma unroll
    for (int ky = 0; ky < 3; ++ky) {
      int iy = oy + ky - 1;
      if (iy < 0 || iy >= 32) continue;
#pragma unroll
      for (int kx = 0; kx < 3; ++kx) {
        int ix = ox + kx - 1;
        if (ix < 0 || ix >= 32) continue;
        acc += x[((bi * 3 + ci) << 10) + (iy << 5) + ix] *
               w[((co * 3 + ci) * 3 + ky) * 3 + kx];
      }
    }
  }
  out[idx] = (_Float16)(acc > 0.f ? acc : 0.f);
}

// ---------------------------------------------------------------------------
// 2x2 max pool, f16 in/out. p = b*C + c plane index.
// ---------------------------------------------------------------------------
__global__ void maxpool2_f16(const _Float16* __restrict__ in,
                             _Float16* __restrict__ out,
                             int Wi, int Ho, int Wo, int total) {
  int i = blockIdx.x * blockDim.x + threadIdx.x;
  if (i >= total) return;
  int x = i % Wo;
  int y = (i / Wo) % Ho;
  int p = i / (Wo * Ho);
  const _Float16* s = in + ((size_t)p * (2 * Ho) + 2 * y) * Wi + 2 * x;
  float m0 = (float)s[0], m1 = (float)s[1];
  float m2 = (float)s[Wi], m3 = (float)s[Wi + 1];
  float m = fmaxf(fmaxf(m0, m1), fmaxf(m2, m3));
  out[i] = (_Float16)m;
}

// ---------------------------------------------------------------------------
// Implicit-GEMM conv via V_WMMA_F32_16X16X32_F16 with double-buffered
// LDS-staged im2col. One block per 16-output-position column tile.
// Pipeline: build(i+1) is issued before compute(i) each chunk, so im2col
// gather latency overlaps the WMMA stream; one barrier per chunk.
// Fragment layouts used by this code (wave32; contiguous k, same in A and B):
//   A lane(l): m=l%16, h=l/16; e0..15 -> K=16h+e
//   B lane(l): n=l%16, h=l/16; e0..15 -> K=16h+e  (two ds_load_b128)
//   D lane(l): n=l%16; vgpr r -> M = r + 8h
// ---------------------------------------------------------------------------
template <int CIN, int KH, int KW, int H, int W, int PAD, int COUT, int KC>
__global__ __launch_bounds__(256) void conv_wmma_lds(
    const _Float16* __restrict__ act,   // [B][CIN][H][W]
    const _Float16* __restrict__ wgt,   // [COUT][K]
    const float* __restrict__ bias,
    _Float16* __restrict__ out) {       // [B][COUT][H][W]
  constexpr int K = CIN * KH * KW;      // multiple of KC
  constexpr int NCHUNK = K / KC;
  constexpr int MT = COUT / 16;
  constexpr int TM = (MT + 7) / 8;      // row tiles per wave
  constexpr int KCP = KC + 8;           // padded LDS row stride (halves)
  constexpr int PANEL = 16 * KCP;
  __shared__ _Float16 smem[2 * PANEL];

  const int tid = threadIdx.x;
  const int lane = tid & 31;
  const int wave = tid >> 5;
  const int nn = lane & 15;
  const int hx = lane >> 4;
  const int tn = blockIdx.x;
  const int hw = H * W;

  // column (output position) decode for this lane (B col / D col)
  const int n = tn * 16 + nn;
  const int bi = n / hw;
  const int s = n % hw;
  const int oy = s / W;
  const int ox = s % W;

  // gather mapping for the build phase: this thread fills column cn
  const int cn = tid & 15;
  const int gn = tn * 16 + cn;
  const int gbi = gn / hw;
  const int gs = gn % hw;
  const int goy = gs / W;
  const int gox = gs % W;

  // A row pointer per row-tile handled by this wave
  const _Float16* Arow[TM];
#pragma unroll
  for (int t = 0; t < TM; ++t) {
    int tm = wave + 8 * t;
    Arow[t] = wgt + (size_t)((tm < MT ? tm : 0) * 16 + nn) * K;
  }

  // build panel `idx` with im2col chunk starting at kc
  auto build = [&](int idx, int kc) {
    _Float16* dst = smem + (idx & 1) * PANEL + cn * KCP;
    for (int kk = tid >> 4; kk < KC; kk += 16) {
      int k = kc + kk;
      int ci = k / (KH * KW);
      int r = k % (KH * KW);
      int iy = goy + r / KW - PAD;
      int ix = gox + r % KW - PAD;
      _Float16 v = (_Float16)0.0f;
      if (iy >= 0 && iy < H && ix >= 0 && ix < W)
        v = act[(size_t)((gbi * CIN + ci) * H + iy) * W + ix];
      dst[kk] = v;
    }
  };

  v8f acc[TM] = {};

  build(0, 0);
  __syncthreads();

  for (int i = 0; i < NCHUNK; ++i) {
    const int kc = i * KC;
    if (i + 1 < NCHUNK) build(i + 1, kc + KC);  // overlap with compute(i)

#pragma unroll
    for (int t = 0; t < TM; ++t)
      __builtin_prefetch(Arow[t] + kc + KC, 0, 1);

    const _Float16* panel = smem + (i & 1) * PANEL;
#pragma unroll 4
    for (int k0 = 0; k0 < KC; k0 += 32) {
      HFrag b;
      const _Float16* bp = panel + nn * KCP + k0 + 16 * hx;
      b.h[0] = *(const v8h*)bp;
      b.h[1] = *(const v8h*)(bp + 8);
#pragma unroll
      for (int t = 0; t < TM; ++t) {
        if (wave + 8 * t < MT) {          // wave-uniform
          HFrag a;
          const _Float16* ap = Arow[t] + kc + k0;
          a.h[0] = *(const v8h*)(ap + 16 * hx);
          a.h[1] = *(const v8h*)(ap + 16 * hx + 8);
          acc[t] = __builtin_amdgcn_wmma_f32_16x16x32_f16(
              false, a.v, false, b.v, (short)0, acc[t], false, false);
        }
      }
    }
    __syncthreads();  // build(i+1) visible; compute(i) done before reuse
  }

#pragma unroll
  for (int t = 0; t < TM; ++t) {
    int tm = wave + 8 * t;
    if (tm < MT) {
#pragma unroll
      for (int r = 0; r < 8; ++r) {
        int mo = tm * 16 + r + 8 * hx;
        float val = acc[t][r] + bias[mo];
        val = val > 0.f ? val : 0.f;
        out[(size_t)((bi * COUT + mo) * H + oy) * W + ox] = (_Float16)val;
      }
    }
  }
}

// ---------------------------------------------------------------------------
// FC GEMM via WMMA with double-buffered LDS activation panel staged through
// GLOBAL_LOAD_ASYNC_TO_LDS_B128 when available (ASYNCcnt / s_wait_asynccnt).
//   act f16 [N][K], wgt f16 [M][K] (torch [out,in]), out[n][m].
// Block = 8 consecutive M-tiles x one N-tile. M padded to 16 with guarded
// loads/stores (fc3: M=1000). Wave-uniform guards keep barriers convergent.
// ---------------------------------------------------------------------------
template <bool RELU, bool F32OUT, int KC>
__global__ __launch_bounds__(256) void fc_wmma_lds(
    const _Float16* __restrict__ act, const _Float16* __restrict__ wgt,
    const float* __restrict__ bias, void* __restrict__ outp,
    int M, int K, int N) {
  constexpr int KCP = KC + 8;
  constexpr int PANEL = 16 * KCP;
  __shared__ _Float16 smem[2 * PANEL];

  const int tid = threadIdx.x;
  const int lane = tid & 31;
  const int wave = tid >> 5;
  const int nn = lane & 15;
  const int hx = lane >> 4;

  const int mt = (M + 15) >> 4;
  const int mg = (mt + 7) >> 3;          // groups of 8 row tiles
  const int gm = blockIdx.x % mg;
  const int tn = blockIdx.x / mg;
  const int tm = gm * 8 + wave;
  const bool tmv = (tm < mt);            // wave-uniform
  const int m = tm * 16 + nn;
  const bool mv = tmv && (m < M);        // lane-level (last tile of fc3)
  const int n = tn * 16 + nn;

  const _Float16* Arow = wgt + (size_t)(mv ? m : 0) * K;

  // copy mapping: threads 16t..16t+15 copy row rn in 16B chunks (coalesced)
  const int rn = tid >> 4;
  const int cb = (tid & 15) * 8;
  const _Float16* srcRow = act + (size_t)(tn * 16 + rn) * K;

  auto buildf = [&](int idx, int kc) {
    _Float16* dst = smem + (idx & 1) * PANEL + rn * KCP;
    const _Float16* src = srcRow + kc;
    for (int kk = cb; kk < KC; kk += 128)
      copy16B_g2l(src + kk, dst + kk);
  };

  const int nchunk = K / KC;
  v8f acc = {};

  buildf(0, 0);
  wait_async_copies();
  __syncthreads();

  for (int i = 0; i < nchunk; ++i) {
    const int kc = i * KC;
    if (i + 1 < nchunk) buildf(i + 1, kc + KC);  // overlap with compute(i)

    if (tmv) {
      __builtin_prefetch(Arow + kc + KC, 0, 1);
      const _Float16* panel = smem + (i & 1) * PANEL;
#pragma unroll 4
      for (int k0 = 0; k0 < KC; k0 += 32) {
        HFrag a, b;
        if (mv) {
          a.h[0] = *(const v8h*)(Arow + kc + k0 + 16 * hx);
          a.h[1] = *(const v8h*)(Arow + kc + k0 + 16 * hx + 8);
        } else {
#pragma unroll
          for (int e = 0; e < 16; ++e) a.e[e] = (_Float16)0.0f;
        }
        const _Float16* bp = panel + nn * KCP + k0 + 16 * hx;
        b.h[0] = *(const v8h*)bp;
        b.h[1] = *(const v8h*)(bp + 8);
        acc = __builtin_amdgcn_wmma_f32_16x16x32_f16(
            false, a.v, false, b.v, (short)0, acc, false, false);
      }
    }
    wait_async_copies();  // panel (i+1) resident before next compute
    __syncthreads();
  }

  if (tmv) {
#pragma unroll
    for (int r = 0; r < 8; ++r) {
      int mo = tm * 16 + r + 8 * hx;
      if (mo >= M) continue;
      float val = acc[r] + bias[mo];
      if (RELU) val = val > 0.f ? val : 0.f;
      if (F32OUT)
        ((float*)outp)[(size_t)n * M + mo] = val;
      else
        ((_Float16*)outp)[(size_t)n * M + mo] = (_Float16)val;
    }
  }
}

// ---------------------------------------------------------------------------
// Launch: sequential layer pipeline on `stream`. Activations stay f16 after
// conv1 in ping-pong buffers P0/P1; weights converted f32->f16 per layer
// into a shared buffer WH (stream-ordered, so reuse is safe).
//   P0: 16,777,216 halves (33.55 MB)   P1: 4,194,304 halves (8.39 MB)
//   WH: 16,777,216 halves (33.55 MB)   total ws: 75,497,472 B
// ---------------------------------------------------------------------------
extern "C" void kernel_launch(void* const* d_in, const int* in_sizes, int n_in,
                              void* d_out, int out_size, void* d_ws,
                              size_t ws_size, hipStream_t stream) {
  (void)in_sizes; (void)n_in; (void)out_size;
  const float* x   = (const float*)d_in[0];
  const float* w1  = (const float*)d_in[1];
  const float* b1  = (const float*)d_in[2];
  const float* w2  = (const float*)d_in[3];
  const float* b2  = (const float*)d_in[4];
  const float* w3  = (const float*)d_in[5];
  const float* b3  = (const float*)d_in[6];
  const float* w4  = (const float*)d_in[7];
  const float* b4  = (const float*)d_in[8];
  const float* w5  = (const float*)d_in[9];
  const float* b5  = (const float*)d_in[10];
  const float* fw1 = (const float*)d_in[11];
  const float* fb1 = (const float*)d_in[12];
  const float* fw2 = (const float*)d_in[13];
  const float* fb2 = (const float*)d_in[14];
  const float* fw3 = (const float*)d_in[15];
  const float* fb3 = (const float*)d_in[16];

  const size_t P0_BYTES = 33554432;  // 16,777,216 halves
  const size_t P1_BYTES = 8388608;   //  4,194,304 halves
  const size_t WH_BYTES = 33554432;
  if (ws_size < P0_BYTES + P1_BYTES + WH_BYTES) return;
  _Float16* P0 = (_Float16*)d_ws;
  _Float16* P1 = (_Float16*)((char*)d_ws + P0_BYTES);
  _Float16* WH = (_Float16*)((char*)d_ws + P0_BYTES + P1_BYTES);

  // conv1 + ReLU -> P0 [256,64,32,32] f16
  conv1_relu_f16<<<65536, 256, 0, stream>>>(x, w1, b1, P0);
  // pool1 -> P1 [256,64,16,16]
  maxpool2_f16<<<(4194304 + 255) / 256, 256, 0, stream>>>(P0, P1, 32, 16, 16,
                                                          4194304);
  // conv2 (K=1600, M=192, N=65536, KC=320) -> P0 [256,192,16,16]
  cvt_f32_f16<<<(307200 + 255) / 256, 256, 0, stream>>>(w2, WH, 307200);
  conv_wmma_lds<64, 5, 5, 16, 16, 2, 192, 320>
      <<<4096, 256, 0, stream>>>(P1, WH, b2, P0);
  // pool2 -> P1 [256,192,8,8]
  maxpool2_f16<<<(3145728 + 255) / 256, 256, 0, stream>>>(P0, P1, 16, 8, 8,
                                                          3145728);
  // conv3 (K=1728, M=384, N=16384, KC=576) -> P0 [256,384,8,8]
  cvt_f32_f16<<<(663552 + 255) / 256, 256, 0, stream>>>(w3, WH, 663552);
  conv_wmma_lds<192, 3, 3, 8, 8, 1, 384, 576>
      <<<1024, 256, 0, stream>>>(P1, WH, b3, P0);
  // conv4 (K=3456, M=256) -> P1 [256,256,8,8]
  cvt_f32_f16<<<(884736 + 255) / 256, 256, 0, stream>>>(w4, WH, 884736);
  conv_wmma_lds<384, 3, 3, 8, 8, 1, 256, 576>
      <<<1024, 256, 0, stream>>>(P0, WH, b4, P1);
  // conv5 (K=2304, M=256) -> P0 [256,256,8,8]
  cvt_f32_f16<<<(589824 + 255) / 256, 256, 0, stream>>>(w5, WH, 589824);
  conv_wmma_lds<256, 3, 3, 8, 8, 1, 256, 576>
      <<<1024, 256, 0, stream>>>(P1, WH, b5, P0);
  // pool5 -> P1 [256,256,4,4] == [256,4096] (NCHW flatten matches reference)
  maxpool2_f16<<<(1048576 + 255) / 256, 256, 0, stream>>>(P0, P1, 8, 4, 4,
                                                          1048576);
  // fc1: [256,4096] x [4096,4096]^T + relu -> P0   (mt=256, mg=32 -> 512 blk)
  cvt_f32_f16<<<65536, 256, 0, stream>>>(fw1, WH, 16777216);
  fc_wmma_lds<true, false, 512>
      <<<512, 256, 0, stream>>>(P1, WH, fb1, P0, 4096, 4096, 256);
  // fc2 -> P1
  cvt_f32_f16<<<65536, 256, 0, stream>>>(fw2, WH, 16777216);
  fc_wmma_lds<true, false, 512>
      <<<512, 256, 0, stream>>>(P0, WH, fb2, P1, 4096, 4096, 256);
  // fc3 -> d_out f32 [256,1000] (mt=63, mg=8 -> 128 blocks)
  cvt_f32_f16<<<(4096000 + 255) / 256, 256, 0, stream>>>(fw3, WH, 4096000);
  fc_wmma_lds<false, true, 512>
      <<<128, 256, 0, stream>>>(P1, WH, fb3, d_out, 1000, 4096, 256);
}
